// GRUPoem_41918880809328
// MI455X (gfx1250) — matrix operand order repaired and checked by reading, not compiled, not run on hardware
//
#include <hip/hip_runtime.h>
#include <hip/hip_bf16.h>

// ---------------------------------------------------------------------------
// GRU LM on MI455X (gfx1250), wave32, v_wmma_f32_16x16x32_bf16 + TDM staging.
//
// Roofline: output projection = 268 GFLOP bf16; 1.05 GB fp32 output stream ->
// ~45us HBM floor @23.3TB/s. W (32.8MB bf16) is L2-resident; block-tiles the
// GEMM 128x128 with the Tensor Data Mover staging W tiles into LDS so total
// L2 traffic drops from ~21GB (per-wave loads) to ~4GB. Recurrent scan is
// sequential/latency-bound -> 1 workgroup, 32 waves, padded LDS state.
// ---------------------------------------------------------------------------

typedef __attribute__((ext_vector_type(16))) __bf16   v16bf;
typedef __attribute__((ext_vector_type(8)))  float    v8f;
typedef __attribute__((ext_vector_type(4)))  unsigned v4u;
typedef __attribute__((ext_vector_type(8)))  int      v8i;
typedef __attribute__((ext_vector_type(4)))  int      v4i;

#define WMMA_BF16(a, b, c) \
    __builtin_amdgcn_wmma_f32_16x16x32_bf16(false, (a), false, (b), (short)0, (c), false, false)

#if defined(__HIP_DEVICE_COMPILE__) && defined(__has_builtin)
# if __has_builtin(__builtin_amdgcn_tensor_load_to_lds) && __has_builtin(__builtin_amdgcn_s_wait_tensorcnt)
#  define HAVE_TDM 1
# endif
#endif
#ifndef HAVE_TDM
# define HAVE_TDM 0
#endif

static constexpr int V = 32000, E = 512, H = 512, S = 128, B = 64;
static constexpr int SB = S * B;                 // 8192 rows
static constexpr size_t OUT_Y = (size_t)SB * V;  // offset of hT in d_out
static constexpr int WROW = H + E;               // 1024, W{r,z,h} row length
static constexpr int HP  = 520;                  // padded bf16 LDS row (1040 B)
static constexpr int HPF = 516;                  // padded f32 LDS row (2064 B)
static constexpr int BROW = 40;                  // k_out LDS B row: 32 data + 4 pad dw (80 B)

__device__ __forceinline__ __bf16 f2bf(float f) {
    unsigned u = __builtin_bit_cast(unsigned, f);
    unsigned short s = (unsigned short)((u + 0x7FFFu + ((u >> 16) & 1u)) >> 16); // RNE
    return __builtin_bit_cast(__bf16, s);
}

__device__ __forceinline__ float sigmoidf_(float x) {
    return 1.0f / (1.0f + __expf(-x));
}

// A-fragment (16x32 bf16): per lane one contiguous 32B run, k = 16*khalf + s (same order as B).
__device__ __forceinline__ v16bf load_a_bf(const __bf16* __restrict__ row, int khalf) {
    v16bf a;
#pragma unroll
    for (int r = 0; r < 8; ++r) {
        int kb = 2 * r + khalf * 16;
        a[2 * r]     = row[kb];
        a[2 * r + 1] = row[kb + 1];
    }
    return a;
}

// B-fragment from 16B-aligned (not necessarily 32B-aligned) memory.
__device__ __forceinline__ v16bf load_b_16al(const __bf16* p) {
    struct alignas(16) U { uint4 lo, hi; } u;
    u.lo = *(const uint4*)p;
    u.hi = *(const uint4*)(p + 8);
    return __builtin_bit_cast(v16bf, u);
}

// ---------------------------------------------------------------------------
__global__ __launch_bounds__(256) void k_cvt_bf16(const float* __restrict__ src,
                                                  __bf16* __restrict__ dst, int n) {
    int i = blockIdx.x * 256 + threadIdx.x;
    if (i < n) dst[i] = f2bf(src[i]);
}

__global__ __launch_bounds__(256) void k_gather(const int* __restrict__ x,
                                                const float* __restrict__ emb,
                                                __bf16* __restrict__ xe) {
    int i = blockIdx.x * 256 + threadIdx.x;      // exactly SB*E threads
    int row = i >> 9, e = i & 511;
    int tok = x[row];
    xe[i] = f2bf(emb[(size_t)tok * E + e]);
}

// ---------------------------------------------------------------------------
// X-projections, all 3 gates per wave-tile (A-frag reused for 3 WMMAs).
__global__ __launch_bounds__(256) void k_xproj(const __bf16* __restrict__ xe,
                                               const __bf16* __restrict__ wr,
                                               const __bf16* __restrict__ wz,
                                               const __bf16* __restrict__ wh,
                                               const float* __restrict__ br,
                                               const float* __restrict__ bz,
                                               const float* __restrict__ bh,
                                               float* __restrict__ xr,
                                               float* __restrict__ xz,
                                               float* __restrict__ xh) {
    int lane = threadIdx.x & 31, wave = threadIdx.x >> 5;
    int tile = blockIdx.x * 8 + wave;            // 512*32 tiles
    int m0 = (tile >> 5) * 16, n0 = (tile & 31) * 16;
    int khalf = lane >> 4, lcol = lane & 15;
    int m = m0 + lcol, n = n0 + lcol;

    v8f cr = {}, cz = {}, ch = {};
    for (int k0 = 0; k0 < E; k0 += 32) {
        v16bf a  = load_a_bf(xe + (size_t)m * E + k0, khalf);
        size_t wo = (size_t)n * WROW + k0 + khalf * 16;
        v16bf b0 = *(const v16bf*)(wr + wo);
        v16bf b1 = *(const v16bf*)(wz + wo);
        v16bf b2 = *(const v16bf*)(wh + wo);
        cr = WMMA_BF16(a, b0, cr);
        cz = WMMA_BF16(a, b1, cz);
        ch = WMMA_BF16(a, b2, ch);
    }
    int nn = n0 + lcol;
    float vr = br[nn], vz = bz[nn], vh = bh[nn];
#pragma unroll
    for (int r = 0; r < 8; ++r) {
        size_t o = (size_t)(m0 + khalf * 8 + r) * H + nn;
        xr[o] = cr[r] + vr;
        xz[o] = cz[r] + vz;
        xh[o] = ch[r] + vh;
    }
}

// ---------------------------------------------------------------------------
// Recurrent scan: 1 workgroup / 32 waves. LDS: h (f32, padded), h mirror
// (bf16, padded), h*r (bf16, padded) -> frag reads hit distinct bank groups.
__global__ __launch_bounds__(1024) void k_rnn(const float* __restrict__ xr,
                                              const float* __restrict__ xz,
                                              const float* __restrict__ xh,
                                              const __bf16* __restrict__ wr,
                                              const __bf16* __restrict__ wz,
                                              const __bf16* __restrict__ wh,
                                              __bf16* __restrict__ hsall,
                                              float* __restrict__ outHT) {
    __shared__ float  hcur[B * HPF];                 // 132 KB fp32 state
    __shared__ __align__(16) __bf16 hbf[B * HP];     // 65 KB bf16 mirror
    __shared__ __align__(16) __bf16 hrb[B * HP];     // 65 KB bf16 h*r

    int tid = threadIdx.x, lane = tid & 31, wave = tid >> 5;
    int khalf = lane >> 4, lcol = lane & 15;
    for (int i = tid; i < B * HPF; i += 1024) hcur[i] = 0.0f;
    for (int i = tid; i < B * HP; i += 1024) hbf[i] = __builtin_bit_cast(__bf16, (unsigned short)0);
    __syncthreads();

    for (int t = 0; t < S; ++t) {
        // ---- r & z gates together: one A-frag, two WMMAs per K step
        v8f zreg[4];
#pragma unroll
        for (int j = 0; j < 4; ++j) {
            int rem = wave + 32 * j;
            int m0 = (rem >> 5) * 16, n0 = (rem & 31) * 16;
            int m = m0 + lcol, n = n0 + lcol;
            v8f cr = {}, cz = {};
            for (int k0 = 0; k0 < H; k0 += 32) {
                v16bf a  = load_a_bf(hbf + m * HP + k0, khalf);
                size_t wo = (size_t)n * WROW + E + k0 + khalf * 16;
                v16bf b0 = *(const v16bf*)(wr + wo);
                v16bf b1 = *(const v16bf*)(wz + wo);
                cr = WMMA_BF16(a, b0, cr);
                cz = WMMA_BF16(a, b1, cz);
            }
#pragma unroll
            for (int r = 0; r < 8; ++r) {
                int mm = m0 + khalf * 8 + r, nn = n0 + lcol;
                size_t xo = (size_t)(t * B + mm) * H + nn;
                float rv = sigmoidf_(cr[r] + xr[xo]);
                hrb[mm * HP + nn] = f2bf(hcur[mm * HPF + nn] * rv);
                zreg[j][r] = sigmoidf_(cz[r] + xz[xo]);
            }
        }
        __syncthreads();   // hrb complete, all hbf/hcur reads done

        // ---- candidate + update: hc = tanh((h*r)@Whh.T + Xh), h = z*h+(1-z)*hc
#pragma unroll
        for (int j = 0; j < 4; ++j) {
            int rem = wave + 32 * j;
            int m0 = (rem >> 5) * 16, n0 = (rem & 31) * 16;
            int m = m0 + lcol, n = n0 + lcol;
            v8f c = {};
            for (int k0 = 0; k0 < H; k0 += 32) {
                v16bf a = load_a_bf(hrb + m * HP + k0, khalf);
                v16bf b = *(const v16bf*)(wh + (size_t)n * WROW + E + k0 + khalf * 16);
                c = WMMA_BF16(a, b, c);
            }
#pragma unroll
            for (int r = 0; r < 8; ++r) {
                int mm = m0 + khalf * 8 + r, nn = n0 + lcol;
                float hc   = tanhf(c[r] + xh[(size_t)(t * B + mm) * H + nn]);
                float z    = zreg[j][r];
                float hnew = z * hcur[mm * HPF + nn] + (1.0f - z) * hc;
                hcur[mm * HPF + nn] = hnew;
                __bf16 hb = f2bf(hnew);
                hbf[mm * HP + nn] = hb;
                hsall[(size_t)(t * B + mm) * H + nn] = hb;
                if (t == S - 1) outHT[nn * B + mm] = hnew;   // hT.T is (H,B)
            }
        }
        __syncthreads();   // state updated before next step
    }
}

// ---------------------------------------------------------------------------
// Output projection: y = Hs @ W.T + b  (8192 x 32000, K=512).
// Block = 8 waves -> 128x128 tile. W tile (128n x 32k) staged in LDS per
// K-chunk via the Tensor Data Mover (double-buffered, pad 16B per 64B row),
// fallback to plain staging if the TDM builtin is unavailable.
__global__ __launch_bounds__(256) void k_out(const __bf16* __restrict__ hs,
                                             const __bf16* __restrict__ wbf,
                                             const float* __restrict__ bo,
                                             float* __restrict__ out) {
    __shared__ __align__(16) __bf16 Bs[2][128 * BROW];   // 2 x 10 KB

    int lane = threadIdx.x & 31, wave = threadIdx.x >> 5;
    int m0 = blockIdx.y * 128 + wave * 16;
    int n0 = blockIdx.x * 128;
    int khalf = lane >> 4, lcol = lane & 15;
    int m = m0 + lcol;

    v8f c[8] = {{}, {}, {}, {}, {}, {}, {}, {}};

#if HAVE_TDM
    // D# groups for a 2D tile: 128 rows x 32 bf16, row stride 512 elems,
    // LDS pad 4 dwords after every 16 dwords (-> 80 B LDS row stride).
    // clang-23 builtin form: (v4u g0, v8i g1, v4i g2, v4i g3, v8i extra, cpol)
    const int g1w0 = (1 << 16)    // data_size = 2 bytes
                   | (1 << 20)    // pad_enable
                   | (3 << 22)    // pad_interval: 16 dwords
                   | (3 << 25);   // pad_amount: 4 dwords
    const v4i gz = {0, 0, 0, 0};
    const v8i gpad = {0, 0, 0, 0, 0, 0, 0, 0};
    if (wave == 0) {
        unsigned lds0 = (unsigned)(uintptr_t)&Bs[0][0];
        unsigned long long ga =
            (unsigned long long)(uintptr_t)(wbf + (size_t)n0 * H);
        v4u g0 = {1u, lds0, (unsigned)(ga & 0xFFFFFFFFu),
                  (unsigned)((ga >> 32) & 0x01FFFFFFu) | (2u << 30)};
        v8i g1 = {g1w0, (int)(512u << 16), (int)((unsigned)V << 16),
                  (int)(32u << 16), 128, 512, 0, 0};
        __builtin_amdgcn_tensor_load_to_lds(g0, g1, gz, gz, gpad, 0);
    }
    for (int kc = 0; kc < 16; ++kc) {
        int p = kc & 1;
        if (wave == 0) __builtin_amdgcn_s_wait_tensorcnt(0);
        __syncthreads();                         // tile kc visible; buf 1-p free
        if (wave == 0 && kc < 15) {
            unsigned lds1 = (unsigned)(uintptr_t)&Bs[1 - p][0];
            unsigned long long ga = (unsigned long long)(uintptr_t)
                (wbf + (size_t)n0 * H + (kc + 1) * 32);
            v4u g0 = {1u, lds1, (unsigned)(ga & 0xFFFFFFFFu),
                      (unsigned)((ga >> 32) & 0x01FFFFFFu) | (2u << 30)};
            v8i g1 = {g1w0, (int)(512u << 16), (int)((unsigned)V << 16),
                      (int)(32u << 16), 128, 512, 0, 0};
            __builtin_amdgcn_tensor_load_to_lds(g0, g1, gz, gz, gpad, 0);
        }
        int k0 = kc * 32;
        v16bf a = load_a_bf(hs + (size_t)m * H + k0, khalf);
#pragma unroll
        for (int j = 0; j < 8; ++j) {
            v16bf b = load_b_16al(&Bs[p][(j * 16 + lcol) * BROW + khalf * 16]);
            c[j] = WMMA_BF16(a, b, c[j]);
        }
    }
#else
    for (int kc = 0; kc < 16; ++kc) {
        int k0 = kc * 32;
        __syncthreads();                         // previous chunk reads done
        int seg = threadIdx.x;                   // 512 x 16B segments, 2/thread
#pragma unroll
        for (int it = 0; it < 2; ++it, seg += 256) {
            int row = seg >> 2, sub = seg & 3;
            uint4 v = *(const uint4*)(wbf + (size_t)(n0 + row) * H + k0 + sub * 8);
            *(uint4*)&Bs[0][row * BROW + sub * 8] = v;
        }
        __syncthreads();
        v16bf a = load_a_bf(hs + (size_t)m * H + k0, khalf);
#pragma unroll
        for (int j = 0; j < 8; ++j) {
            v16bf b = load_b_16al(&Bs[0][(j * 16 + lcol) * BROW + khalf * 16]);
            c[j] = WMMA_BF16(a, b, c[j]);
        }
    }
#endif

#pragma unroll
    for (int j = 0; j < 8; ++j) {
        int nn = n0 + j * 16 + lcol;
        float bb = bo[nn];
#pragma unroll
        for (int r = 0; r < 8; ++r) {
            out[(size_t)(m0 + khalf * 8 + r) * V + nn] = c[j][r] + bb;
        }
    }
}

// ---------------------------------------------------------------------------
extern "C" void kernel_launch(void* const* d_in, const int* in_sizes, int n_in,
                              void* d_out, int out_size, void* d_ws, size_t ws_size,
                              hipStream_t stream) {
    const int*   x   = (const int*)  d_in[0];
    const float* emb = (const float*)d_in[1];
    const float* Wr  = (const float*)d_in[2];
    const float* Br  = (const float*)d_in[3];
    const float* Wz  = (const float*)d_in[4];
    const float* Bz  = (const float*)d_in[5];
    const float* Wh  = (const float*)d_in[6];
    const float* Bh  = (const float*)d_in[7];
    const float* W   = (const float*)d_in[8];
    const float* b   = (const float*)d_in[9];
    float* out = (float*)d_out;

    char* ws = (char*)d_ws;
    size_t off = 0;
    __bf16* xe  = (__bf16*)(ws + off); off += (size_t)SB * E * 2;
    __bf16* wrb = (__bf16*)(ws + off); off += (size_t)H * WROW * 2;
    __bf16* wzb = (__bf16*)(ws + off); off += (size_t)H * WROW * 2;
    __bf16* whb = (__bf16*)(ws + off); off += (size_t)H * WROW * 2;
    __bf16* wbf = (__bf16*)(ws + off); off += (size_t)V * H * 2;
    float*  xrp = (float*) (ws + off); off += (size_t)SB * H * 4;
    float*  xzp = (float*) (ws + off); off += (size_t)SB * H * 4;
    float*  xhp = (float*) (ws + off); off += (size_t)SB * H * 4;
    __bf16* hsb = (__bf16*)(ws + off); off += (size_t)SB * H * 2;

    const int nWg = H * WROW;
    k_cvt_bf16<<<(nWg + 255) / 256, 256, 0, stream>>>(Wr, wrb, nWg);
    k_cvt_bf16<<<(nWg + 255) / 256, 256, 0, stream>>>(Wz, wzb, nWg);
    k_cvt_bf16<<<(nWg + 255) / 256, 256, 0, stream>>>(Wh, whb, nWg);
    k_cvt_bf16<<<(V * H + 255) / 256, 256, 0, stream>>>(W, wbf, V * H);

    k_gather<<<SB * E / 256, 256, 0, stream>>>(x, emb, xe);

    k_xproj<<<2048, 256, 0, stream>>>(xe, wrb, wzb, whb, Br, Bz, Bh, xrp, xzp, xhp);

    k_rnn<<<1, 1024, 0, stream>>>(xrp, xzp, xhp, wrb, wzb, whb, hsb, out + OUT_Y);

    k_out<<<dim3(V / 128, SB / 128), 256, 0, stream>>>(hsb, wbf, b, out);
}
